// LinearAttentionKernel_47785806135913
// MI455X (gfx1250) — matrix operand order repaired and checked by reading, not compiled, not run on hardware
//
#include <hip/hip_runtime.h>
#include <hip/hip_bf16.h>

// ---------------------------------------------------------------------------
// Types for CDNA5 WMMA (wave32)
// ---------------------------------------------------------------------------
typedef __bf16 bf16_t;
typedef __attribute__((ext_vector_type(16))) __bf16 v16bf;
typedef __attribute__((ext_vector_type(8)))  __bf16 v8bf;
typedef __attribute__((ext_vector_type(8)))  float  v8f;

__device__ __forceinline__ float bf2f(bf16_t h) {
    unsigned short s = __builtin_bit_cast(unsigned short, h);
    unsigned u = ((unsigned)s) << 16;
    return __builtin_bit_cast(float, u);
}
__device__ __forceinline__ bf16_t f2bf(float f) {
    unsigned u = __builtin_bit_cast(unsigned, f);
    u = u + 0x7FFFu + ((u >> 16) & 1u);           // round-to-nearest-even
    unsigned short s = (unsigned short)(u >> 16);
    return __builtin_bit_cast(bf16_t, s);
}
__device__ __forceinline__ v8f zero_v8f() {
    v8f z;
#pragma unroll
    for (int i = 0; i < 8; ++i) z[i] = 0.f;
    return z;
}

// A operand (16x32 bf16, MxK). Row-major source with leading dim `lda`.
// lane<16: M=lane, K={0..7,16..23}; lane>=16: M=lane-16, K={8..15,24..31}.
__device__ __forceinline__ v16bf load_a_frag(const bf16_t* __restrict__ base,
                                             int row0, int lda, int k0, int lane) {
    int l  = lane & 15;
    int hi = lane >> 4;
    const bf16_t* p = base + (size_t)(row0 + l) * lda + k0 + hi * 8;
    v8bf lo = *(const v8bf*)p;
    v8bf hh = *(const v8bf*)(p + 16);
    return __builtin_shufflevector(lo, hh, 0,1,2,3,4,5,6,7,8,9,10,11,12,13,14,15);
}

// B operand (32x16 bf16, KxN). Source is N-major rows with contiguous K (ldb).
// lane<16: col=lane, K={0..7,16..23}; lane>=16: col=lane-16, K={8..15,24..31}.
__device__ __forceinline__ v16bf load_b_frag(const bf16_t* __restrict__ base,
                                             int col0, int ldb, int k0, int lane) {
    int l  = lane & 15;
    int hi = lane >> 4;
    const bf16_t* p = base + (size_t)(col0 + l) * ldb + k0 + hi * 8;
    v8bf lo = *(const v8bf*)p;
    v8bf hh = *(const v8bf*)(p + 16);
    return __builtin_shufflevector(lo, hh, 0,1,2,3,4,5,6,7,8,9,10,11,12,13,14,15);
}

__device__ __forceinline__ v8f wmma_bf16(v16bf a, v16bf b, v8f c) {
    return __builtin_amdgcn_wmma_f32_16x16x32_bf16(false, a, false, b,
                                                   (short)0, c, false, false);
}

// ---------------------------------------------------------------------------
// Constants
// ---------------------------------------------------------------------------
#define BATCH 4
#define NTOK 4096
#define CDIM 1024
#define HEADS 16
#define DHEAD 64
#define MROWS (BATCH * NTOK)       // 16384
#define QKVN  (3 * CDIM)           // 3072
#define SCALE 0.3535533905932738f  // 64^-0.25
#define EPS   1e-6f

// ---------------------------------------------------------------------------
// Kernel 0: f32 -> bf16 conversion
// ---------------------------------------------------------------------------
__global__ __launch_bounds__(256)
void cvt_f32_bf16_kernel(const float* __restrict__ src, bf16_t* __restrict__ dst, int n) {
    int i = blockIdx.x * 256 + threadIdx.x;
    if (i < n) dst[i] = f2bf(src[i]);
}

// ---------------------------------------------------------------------------
// Kernel 1: qkv = x @ w_qkv^T  (16384 x 3072 x 1024), epilogue applies the
// feature map and scatters q -> [B,H,N,D], k,v -> transposed [B,H,D,N].
// Block = 128 thr (4 waves), tile = 128x128, wave = 64x64 (4x4 frags):
// 16 WMMAs per 8 operand-fragment loads per k-step.
// ---------------------------------------------------------------------------
__global__ __launch_bounds__(128)
void qkv_gemm_kernel(const bf16_t* __restrict__ X, const bf16_t* __restrict__ W,
                     bf16_t* __restrict__ qf, bf16_t* __restrict__ kT,
                     bf16_t* __restrict__ vT) {
    const int K = CDIM;
    int tid  = threadIdx.x;
    int lane = tid & 31;
    int wave = tid >> 5;
    int wrow = blockIdx.x * 128 + (wave >> 1) * 64;
    int wcol = blockIdx.y * 128 + (wave & 1) * 64;

    v8f acc[4][4];
#pragma unroll
    for (int i = 0; i < 4; ++i)
#pragma unroll
        for (int j = 0; j < 4; ++j) acc[i][j] = zero_v8f();

    for (int k0 = 0; k0 < K; k0 += 32) {
        v16bf a[4], b[4];
#pragma unroll
        for (int i = 0; i < 4; ++i) a[i] = load_a_frag(X, wrow + i * 16, K, k0, lane);
#pragma unroll
        for (int j = 0; j < 4; ++j) b[j] = load_b_frag(W, wcol + j * 16, K, k0, lane);
#pragma unroll
        for (int i = 0; i < 4; ++i)
#pragma unroll
            for (int j = 0; j < 4; ++j)
                acc[i][j] = wmma_bf16(a[i], b[j], acc[i][j]);
    }

    int l  = lane & 15;
    int hi = lane >> 4;
#pragma unroll
    for (int i = 0; i < 4; ++i) {
#pragma unroll
        for (int j = 0; j < 4; ++j) {
#pragma unroll
            for (int r = 0; r < 8; ++r) {
                int m = wrow + i * 16 + r + 8 * hi;   // token row
                int f = wcol + j * 16 + l;            // feature col [0,3072)
                float val = acc[i][j][r];
                int b_ = m >> 12;
                int n = m & (NTOK - 1);
                int which = f >> 10;                  // 0=q, 1=k, 2=v
                int c = f & (CDIM - 1);
                int h = c >> 6;
                int d = c & (DHEAD - 1);
                size_t bh = (size_t)(b_ * HEADS + h);
                if (which == 0) {
                    float fv = val > 0.f ? val + 1.f : __expf(val);
                    qf[(bh * NTOK + n) * DHEAD + d] = f2bf(fv * SCALE);
                } else if (which == 1) {
                    float fv = val > 0.f ? val + 1.f : __expf(val);
                    kT[(bh * DHEAD + d) * NTOK + n] = f2bf(fv * SCALE);
                } else {
                    vT[(bh * DHEAD + d) * NTOK + n] = f2bf(val);
                }
            }
        }
    }
}

// ---------------------------------------------------------------------------
// Kernel 2: per (b,h): kv[d,e] = sum_n k[n,d]*v[n,e]  (64x64x4096 GEMM) plus
// k_sum[d]. kv written transposed (bf16) for the next stage's B operand.
// Block = 128 thr (4 waves), each wave: 16 rows x 64 cols (4 frags).
// ---------------------------------------------------------------------------
__global__ __launch_bounds__(128)
void kv_kernel(const bf16_t* __restrict__ kT, const bf16_t* __restrict__ vT,
               bf16_t* __restrict__ kvT, float* __restrict__ ksum) {
    int bh = blockIdx.x;                       // 0..63
    const bf16_t* kb = kT + (size_t)bh * DHEAD * NTOK;
    const bf16_t* vb = vT + (size_t)bh * DHEAD * NTOK;
    int tid = threadIdx.x, lane = tid & 31, wave = tid >> 5;
    int wrow = wave * 16;

    v8f acc[4];
#pragma unroll
    for (int j = 0; j < 4; ++j) acc[j] = zero_v8f();

    for (int k0 = 0; k0 < NTOK; k0 += 32) {
        v16bf a = load_a_frag(kb, wrow, NTOK, k0, lane);
#pragma unroll
        for (int j = 0; j < 4; ++j) {
            v16bf b = load_b_frag(vb, j * 16, NTOK, k0, lane);
            acc[j] = wmma_bf16(a, b, acc[j]);
        }
    }

    int l = lane & 15, hi = lane >> 4;
#pragma unroll
    for (int j = 0; j < 4; ++j)
#pragma unroll
        for (int r = 0; r < 8; ++r) {
            int d = wrow + r + 8 * hi;
            int e = j * 16 + l;
            kvT[(size_t)bh * DHEAD * DHEAD + (size_t)e * DHEAD + d] = f2bf(acc[j][r]);
        }

    // k_sum[d] = sum_n k[n,d] : one row of kT per thread (threads 0..63)
    if (tid < DHEAD) {
        const bf16_t* p = kb + (size_t)tid * NTOK;
        float s = 0.f;
        for (int n = 0; n < NTOK; n += 8) {
            v8bf chunk = *(const v8bf*)(p + n);
#pragma unroll
            for (int t = 0; t < 8; ++t) s += bf2f(chunk[t]);
        }
        ksum[bh * DHEAD + tid] = s;
    }
}

// ---------------------------------------------------------------------------
// Kernel 3: per (b,h, 64-token tile): out = q @ kv, denom = q . k_sum,
// normalize, write bf16 attn in token-major [B,N,C].
// Block = 128 thr (4 waves), wave = 16 token rows x 64 cols.
// ---------------------------------------------------------------------------
__global__ __launch_bounds__(128)
void attn_out_kernel(const bf16_t* __restrict__ qf, const bf16_t* __restrict__ kvT,
                     const float* __restrict__ ksum, bf16_t* __restrict__ attn) {
    int bh   = blockIdx.y;                     // 0..63
    int tile = blockIdx.x;                     // 0..63 (token tiles of 64)
    int b = bh >> 4, h = bh & 15;
    const bf16_t* qb  = qf  + (size_t)bh * NTOK * DHEAD;
    const bf16_t* kvb = kvT + (size_t)bh * DHEAD * DHEAD;
    const float*  ks  = ksum + bh * DHEAD;

    __shared__ float denom_s[64];
    int tid = threadIdx.x, lane = tid & 31, wave = tid >> 5;
    int row0 = tile * 64;

    if (tid < 64) {
        const bf16_t* qp = qb + (size_t)(row0 + tid) * DHEAD;
        float s = 0.f;
#pragma unroll
        for (int d = 0; d < DHEAD; d += 8) {
            v8bf chunk = *(const v8bf*)(qp + d);
#pragma unroll
            for (int t = 0; t < 8; ++t) s += bf2f(chunk[t]) * ks[d + t];
        }
        denom_s[tid] = s + EPS;
    }
    __syncthreads();

    int wrow = row0 + wave * 16;
    v8f acc[4];
#pragma unroll
    for (int j = 0; j < 4; ++j) acc[j] = zero_v8f();

#pragma unroll
    for (int k0 = 0; k0 < DHEAD; k0 += 32) {
        v16bf a = load_a_frag(qb, wrow, DHEAD, k0, lane);
#pragma unroll
        for (int j = 0; j < 4; ++j) {
            v16bf bfr = load_b_frag(kvb, j * 16, DHEAD, k0, lane);
            acc[j] = wmma_bf16(a, bfr, acc[j]);
        }
    }

    int l = lane & 15, hi = lane >> 4;
#pragma unroll
    for (int j = 0; j < 4; ++j)
#pragma unroll
        for (int r = 0; r < 8; ++r) {
            int mlocal = wave * 16 + r + 8 * hi;     // 0..63 within block
            int n = row0 + mlocal;
            int e = j * 16 + l;
            float val = acc[j][r] / denom_s[mlocal];
            attn[((size_t)b * NTOK + n) * CDIM + h * DHEAD + e] = f2bf(val);
        }
}

// ---------------------------------------------------------------------------
// Kernel 4: out = attn @ w_out^T + b_out  (16384 x 1024 x 1024), f32 output.
// Same 4-wave 64x64-per-wave tiling as kernel 1.
// ---------------------------------------------------------------------------
__global__ __launch_bounds__(128)
void out_gemm_kernel(const bf16_t* __restrict__ A, const bf16_t* __restrict__ W,
                     const float* __restrict__ bias, float* __restrict__ out) {
    const int K = CDIM;
    int tid  = threadIdx.x;
    int lane = tid & 31;
    int wave = tid >> 5;
    int wrow = blockIdx.x * 128 + (wave >> 1) * 64;
    int wcol = blockIdx.y * 128 + (wave & 1) * 64;

    v8f acc[4][4];
#pragma unroll
    for (int i = 0; i < 4; ++i)
#pragma unroll
        for (int j = 0; j < 4; ++j) acc[i][j] = zero_v8f();

    for (int k0 = 0; k0 < K; k0 += 32) {
        v16bf a[4], b[4];
#pragma unroll
        for (int i = 0; i < 4; ++i) a[i] = load_a_frag(A, wrow + i * 16, K, k0, lane);
#pragma unroll
        for (int j = 0; j < 4; ++j) b[j] = load_b_frag(W, wcol + j * 16, K, k0, lane);
#pragma unroll
        for (int i = 0; i < 4; ++i)
#pragma unroll
            for (int j = 0; j < 4; ++j)
                acc[i][j] = wmma_bf16(a[i], b[j], acc[i][j]);
    }

    int l = lane & 15, hi = lane >> 4;
#pragma unroll
    for (int i = 0; i < 4; ++i)
#pragma unroll
        for (int j = 0; j < 4; ++j)
#pragma unroll
            for (int r = 0; r < 8; ++r) {
                int m = wrow + i * 16 + r + 8 * hi;
                int f = wcol + j * 16 + l;
                out[(size_t)m * CDIM + f] = acc[i][j][r] + bias[f];
            }
}

// ---------------------------------------------------------------------------
// Host launcher
// ---------------------------------------------------------------------------
extern "C" void kernel_launch(void* const* d_in, const int* in_sizes, int n_in,
                              void* d_out, int out_size, void* d_ws, size_t ws_size,
                              hipStream_t stream) {
    const float* x     = (const float*)d_in[0];   // [4,4096,1024]
    const float* w_qkv = (const float*)d_in[1];   // [3072,1024]
    const float* w_out = (const float*)d_in[2];   // [1024,1024]
    const float* b_out = (const float*)d_in[3];   // [1024]

    char* ws = (char*)d_ws;
    size_t off = 0;
    auto wsalloc = [&](size_t bytes) -> void* {
        void* p = (void*)(ws + off);
        off += (bytes + 255) & ~(size_t)255;
        return p;
    };

    bf16_t* xb   = (bf16_t*)wsalloc((size_t)MROWS * CDIM * 2);      // 32 MB
    bf16_t* wqb  = (bf16_t*)wsalloc((size_t)QKVN * CDIM * 2);       //  6 MB
    bf16_t* wob  = (bf16_t*)wsalloc((size_t)CDIM * CDIM * 2);       //  2 MB
    bf16_t* qf   = (bf16_t*)wsalloc((size_t)MROWS * CDIM * 2);      // 32 MB
    bf16_t* kT   = (bf16_t*)wsalloc((size_t)MROWS * CDIM * 2);      // 32 MB
    bf16_t* vT   = (bf16_t*)wsalloc((size_t)MROWS * CDIM * 2);      // 32 MB
    bf16_t* kvT  = (bf16_t*)wsalloc((size_t)BATCH * HEADS * DHEAD * DHEAD * 2);
    float*  ksum = (float*) wsalloc((size_t)BATCH * HEADS * DHEAD * 4);
    bf16_t* attn = (bf16_t*)wsalloc((size_t)MROWS * CDIM * 2);      // 32 MB

    // 0) convert inputs to bf16
    {
        int n = MROWS * CDIM;
        cvt_f32_bf16_kernel<<<(n + 255) / 256, 256, 0, stream>>>(x, xb, n);
        n = QKVN * CDIM;
        cvt_f32_bf16_kernel<<<(n + 255) / 256, 256, 0, stream>>>(w_qkv, wqb, n);
        n = CDIM * CDIM;
        cvt_f32_bf16_kernel<<<(n + 255) / 256, 256, 0, stream>>>(w_out, wob, n);
    }

    // 1) QKV GEMM + feature map + layout scatter
    qkv_gemm_kernel<<<dim3(MROWS / 128, QKVN / 128), 128, 0, stream>>>(xb, wqb, qf, kT, vT);

    // 2) per-head kv = k^T v and k_sum
    kv_kernel<<<dim3(BATCH * HEADS), 128, 0, stream>>>(kT, vT, kvT, ksum);

    // 3) per-head out = q kv / (q . k_sum + eps)
    attn_out_kernel<<<dim3(NTOK / 64, BATCH * HEADS), 128, 0, stream>>>(qf, kvT, ksum, attn);

    // 4) output projection + bias
    out_gemm_kernel<<<dim3(MROWS / 128, CDIM / 128), 128, 0, stream>>>(attn, wob, b_out, (float*)d_out);
}
